// SelfAttentionV2_50757923504266
// MI455X (gfx1250) — compile-verified, matrix operand order checked by reading, not run on hardware
//
#include <hip/hip_runtime.h>

// ---------------------------------------------------------------------------
// MI455X (gfx1250) beam-cached attention.
// Dominant cost: K/V projections over the 8 dedup batch rows:
//   2 x (16384 x 1024 x 1024) GEMM = 68.7 GFLOP  -> v_wmma_f32_16x16x32_bf16.
// f32->bf16 hoisted to one-time prepasses; all WMMA fragment loads are 2x b128
// (fragment K-runs are contiguous); V cache stored transposed so P*V fragments
// are vectorizable too; GEMM K-loop software-pipelined across the LDS barriers.
// ---------------------------------------------------------------------------

#define EMBED   1024
#define HEADS   16
#define HDIM    64
#define BEAMS   4
#define CBATCH  8
#define BATCH   32
#define TLEN    4
#define SLEN    2048
#define QSCALE  0.125f   // 64^-0.5

typedef __attribute__((ext_vector_type(16))) __bf16 v16bf;
typedef __attribute__((ext_vector_type(8)))  float  v8f;

union Frag { uint4 q[2]; unsigned int u[8]; v16bf v; };

__device__ __forceinline__ v8f zerov() {
    v8f z;
#pragma unroll
    for (int i = 0; i < 8; ++i) z[i] = 0.f;
    return z;
}

// f32 -> bf16 (round-to-nearest-even)
__device__ __forceinline__ unsigned short f2bf(float x) {
    union { float f; unsigned int u; } c; c.f = x;
    unsigned int r = c.u + 0x7FFFu + ((c.u >> 16) & 1u);
    return (unsigned short)(r >> 16);
}

__device__ __forceinline__ unsigned int pack2(float x, float y) {
    return (unsigned int)f2bf(x) | ((unsigned int)f2bf(y) << 16);
}

// 16-bit WMMA A/B fragment from a row-major [16 rows][ld] bf16 tile:
// lane needs row r = row0 + (lane&15) and two contiguous 8-elem K-runs
// (k = g*8..+7 and 16+g*8..+7, g = lane>>4)  ->  two b128 loads.
// Requires 16B-aligned base and ld % 8 == 0.
__device__ __forceinline__ v16bf load_frag_rowmajor(const unsigned short* t, int row0, int ld, int lane) {
    int r = row0 + (lane & 15);
    int g = lane >> 4;
    const unsigned short* p = t + r * ld + g * 8;
    Frag f;
    f.q[0] = *(const uint4*)(p);        // k = g*8 .. g*8+7
    f.q[1] = *(const uint4*)(p + 16);   // k = 16+g*8 .. 16+g*8+7
    return f.v;
}

// ---------------------------------------------------------------------------
// Prepass 1: gather the 8 dedup key rows -> compact bf16 [m][1024], m = cbi*SLEN + s.
// ---------------------------------------------------------------------------
__global__ __launch_bounds__(256) void cvt_gather_key(const float* __restrict__ key,
                                                      unsigned short* __restrict__ out) {
    long idx = ((long)blockIdx.x * 256 + threadIdx.x) * 8;   // into [16384][1024]
    int m = (int)(idx >> 10);
    int c = (int)(idx & 1023);
    long src = ((long)(m & (SLEN - 1)) * BATCH + (m >> 11) * BEAMS) * EMBED + c;
    float4 f0 = *(const float4*)(key + src);
    float4 f1 = *(const float4*)(key + src + 4);
    uint4 o;
    o.x = pack2(f0.x, f0.y); o.y = pack2(f0.z, f0.w);
    o.z = pack2(f1.x, f1.y); o.w = pack2(f1.z, f1.w);
    *(uint4*)(out + idx) = o;
}

// Prepass 2: plain f32 -> bf16, 8 elements/thread.
__global__ __launch_bounds__(256) void cvt_plain(const float* __restrict__ in,
                                                 unsigned short* __restrict__ out) {
    long idx = ((long)blockIdx.x * 256 + threadIdx.x) * 8;
    float4 f0 = *(const float4*)(in + idx);
    float4 f1 = *(const float4*)(in + idx + 4);
    uint4 o;
    o.x = pack2(f0.x, f0.y); o.y = pack2(f0.z, f0.w);
    o.z = pack2(f1.x, f1.y); o.w = pack2(f1.z, f1.w);
    *(uint4*)(out + idx) = o;
}

// ---------------------------------------------------------------------------
// 128x128-tile GEMM, bf16 in (pre-converted), f32 accum: out = A @ W^T + bias.
// MODE 0: K proj -> bf16 out, layout [cb][h][s][hd]     (A row m = cbi*SLEN+s)
// MODE 1: Q proj -> bf16 out, layout [cb][h][beam*T+t][hd], * QSCALE
// MODE 2: O proj -> f32 out [T*B][E]
// MODE 3: V proj -> bf16 out, TRANSPOSED layout [cb][h][hd][s]
// ---------------------------------------------------------------------------
template<int MODE>
__global__ __launch_bounds__(256) void gemm_wmma(const unsigned short* __restrict__ A,
                                                 const unsigned short* __restrict__ W,
                                                 const float* __restrict__ bias,
                                                 void* __restrict__ outp) {
    __shared__ uint4 AsRaw[128 * 32 / 8];   // 8 KB, 16B-aligned
    __shared__ uint4 BsRaw[128 * 32 / 8];
    unsigned short* As = (unsigned short*)AsRaw;
    unsigned short* Bs = (unsigned short*)BsRaw;

    const int tid  = threadIdx.x;
    const int lane = tid & 31;
    const int wid  = tid >> 5;
    const int wm   = wid & 3;   // 4 waves along M, 32 rows each
    const int wn   = wid >> 2;  // 2 waves along N, 64 cols each
    const int m0   = blockIdx.y * 128;
    const int n0   = blockIdx.x * 128;

    v8f acc[2][4];
#pragma unroll
    for (int a = 0; a < 2; ++a)
#pragma unroll
        for (int b = 0; b < 4; ++b) acc[a][b] = zerov();

    const int srow = tid >> 1;            // 0..127
    const int scol = (tid & 1) * 16;      // 0 or 16

    const unsigned short* Arow = A + (long)(m0 + srow) * EMBED + scol;
    const unsigned short* Wrow = W + (long)(n0 + srow) * EMBED + scol;

    // software pipeline: hold next K-tile in registers across the barrier
    uint4 a0 = *(const uint4*)(Arow);
    uint4 a1 = *(const uint4*)(Arow + 8);
    uint4 b0 = *(const uint4*)(Wrow);
    uint4 b1 = *(const uint4*)(Wrow + 8);

    for (int kk = 0; kk < EMBED; kk += 32) {
        *(uint4*)(&As[srow * 32 + scol])     = a0;
        *(uint4*)(&As[srow * 32 + scol + 8]) = a1;
        *(uint4*)(&Bs[srow * 32 + scol])     = b0;
        *(uint4*)(&Bs[srow * 32 + scol + 8]) = b1;
        __syncthreads();

        int kn = (kk + 32 < EMBED) ? kk + 32 : kk;   // last iter: reload same (discarded)
        a0 = *(const uint4*)(Arow + kn);
        a1 = *(const uint4*)(Arow + kn + 8);
        b0 = *(const uint4*)(Wrow + kn);
        b1 = *(const uint4*)(Wrow + kn + 8);

        v16bf fa0 = load_frag_rowmajor(As, wm * 32,      32, lane);
        v16bf fa1 = load_frag_rowmajor(As, wm * 32 + 16, 32, lane);
#pragma unroll
        for (int j = 0; j < 4; ++j) {
            v16bf fb = load_frag_rowmajor(Bs, wn * 64 + j * 16, 32, lane);
            acc[0][j] = __builtin_amdgcn_wmma_f32_16x16x32_bf16(false, fa0, false, fb,
                                                                (short)0, acc[0][j], false, false);
            acc[1][j] = __builtin_amdgcn_wmma_f32_16x16x32_bf16(false, fa1, false, fb,
                                                                (short)0, acc[1][j], false, false);
        }
        __syncthreads();
    }

    // Epilogue. C layout: element(i, lane): row = i + 8*(lane>=16), col = lane&15.
#pragma unroll
    for (int sub = 0; sub < 2; ++sub)
#pragma unroll
        for (int j = 0; j < 4; ++j)
#pragma unroll
            for (int i = 0; i < 8; ++i) {
                int gm = m0 + wm * 32 + sub * 16 + i + ((lane >> 4) << 3);
                int gn = n0 + wn * 64 + j * 16 + (lane & 15);
                float c = acc[sub][j][i] + bias[gn];
                if (MODE == 0) {
                    unsigned short* o = (unsigned short*)outp;
                    int cbi = gm >> 11, s = gm & (SLEN - 1);
                    int h = gn / HDIM, hd = gn % HDIM;
                    o[(((long)(cbi * HEADS + h)) * SLEN + s) * HDIM + hd] = f2bf(c);
                } else if (MODE == 3) {
                    unsigned short* o = (unsigned short*)outp;
                    int cbi = gm >> 11, s = gm & (SLEN - 1);
                    int h = gn / HDIM, hd = gn % HDIM;
                    o[(((long)(cbi * HEADS + h)) * HDIM + hd) * SLEN + s] = f2bf(c);
                } else if (MODE == 1) {
                    unsigned short* o = (unsigned short*)outp;
                    int t = gm / BATCH, bb = gm % BATCH;
                    int cbi = bb / BEAMS, beam = bb % BEAMS;
                    int h = gn / HDIM, hd = gn % HDIM;
                    int row16 = beam * TLEN + t;
                    o[((cbi * HEADS + h) * 16 + row16) * HDIM + hd] = f2bf(c * QSCALE);
                } else {
                    float* o = (float*)outp;
                    o[(long)gm * EMBED + gn] = c;
                }
            }
}

// ---------------------------------------------------------------------------
// Flash-style attention: one WG per (head, cb). Q tile 16x64 (rows = beam*T+t).
// 8 waves each own a 256-wide S strip; online softmax; cross-wave combine in LDS.
// V is pre-transposed [cb][h][hd][s] so every fragment load is 2x b128.
// Output stored as bf16 for the O-projection GEMM.
// ---------------------------------------------------------------------------
__global__ __launch_bounds__(256) void attn_wmma(const unsigned short* __restrict__ qb,
                                                 const unsigned short* __restrict__ kb,
                                                 const unsigned short* __restrict__ vbT,
                                                 const unsigned char*  __restrict__ kpm,
                                                 unsigned short* __restrict__ ao) {
    __shared__ uint4 pbufRaw[8][16 * 32 / 8];             // per-wave P staging, 16B-aligned
    __shared__ float redM[8][16], redL[8][16];
    __shared__ float redO[8][16][64];
    __shared__ float scaleW[8][16], Lrow[16];

    const int h   = blockIdx.x;
    const int cb  = blockIdx.y;
    const int tid = threadIdx.x;
    const int lane = tid & 31;
    const int w   = tid >> 5;
    const int g   = lane >> 4;
    const int ln  = lane & 15;
    unsigned short* pb = (unsigned short*)pbufRaw[w];

    const unsigned short* qh = qb  + (long)((cb * HEADS + h) * 16) * HDIM;
    const unsigned short* kh = kb  + (long)(cb * HEADS + h) * SLEN * HDIM;
    const unsigned short* vh = vbT + (long)(cb * HEADS + h) * HDIM * SLEN;

    v16bf qlo = load_frag_rowmajor(qh,      0, HDIM, lane);   // k = 0..31
    v16bf qhi = load_frag_rowmajor(qh + 32, 0, HDIM, lane);   // k = 32..63

    v8f acc[4];
#pragma unroll
    for (int j = 0; j < 4; ++j) acc[j] = zerov();
    float m_run[8], l_run[8];
#pragma unroll
    for (int i = 0; i < 8; ++i) { m_run[i] = -3.0e38f; l_run[i] = 0.f; }

    for (int ch = 0; ch < 8; ++ch) {
        int sb = w * 256 + ch * 32;

        v8f s0 = zerov(), s1 = zerov();
        s0 = __builtin_amdgcn_wmma_f32_16x16x32_bf16(false, qlo, false,
                 load_frag_rowmajor(kh + (long)sb * HDIM, 0, HDIM, lane), (short)0, s0, false, false);
        s0 = __builtin_amdgcn_wmma_f32_16x16x32_bf16(false, qhi, false,
                 load_frag_rowmajor(kh + (long)sb * HDIM + 32, 0, HDIM, lane), (short)0, s0, false, false);
        s1 = __builtin_amdgcn_wmma_f32_16x16x32_bf16(false, qlo, false,
                 load_frag_rowmajor(kh + (long)(sb + 16) * HDIM, 0, HDIM, lane), (short)0, s1, false, false);
        s1 = __builtin_amdgcn_wmma_f32_16x16x32_bf16(false, qhi, false,
                 load_frag_rowmajor(kh + (long)(sb + 16) * HDIM + 32, 0, HDIM, lane), (short)0, s1, false, false);

        float mloc[8];
        int msk0[8], msk1[8];
#pragma unroll
        for (int i = 0; i < 8; ++i) {
            int row = i + (g << 3);
            int b = cb * BEAMS + (row >> 2);         // beam = row / TLEN
            int c0 = sb + ln;
            msk0[i] = kpm[(long)b * SLEN + c0];
            msk1[i] = kpm[(long)b * SLEN + c0 + 16];
            float v0 = msk0[i] ? -3.0e38f : s0[i];
            float v1 = msk1[i] ? -3.0e38f : s1[i];
            s0[i] = v0; s1[i] = v1;
            mloc[i] = fmaxf(v0, v1);
        }
#pragma unroll
        for (int i = 0; i < 8; ++i)
#pragma unroll
            for (int off = 1; off < 16; off <<= 1)
                mloc[i] = fmaxf(mloc[i], __shfl_xor(mloc[i], off, 16));

#pragma unroll
        for (int i = 0; i < 8; ++i) {
            int row = i + (g << 3);
            float mnew  = fmaxf(m_run[i], mloc[i]);
            float alpha = (m_run[i] > -1.0e37f) ? __expf(m_run[i] - mnew) : 0.f;
            float p0 = msk0[i] ? 0.f : __expf(s0[i] - mnew);
            float p1 = msk1[i] ? 0.f : __expf(s1[i] - mnew);
            float rs = p0 + p1;
#pragma unroll
            for (int off = 1; off < 16; off <<= 1)
                rs += __shfl_xor(rs, off, 16);
            l_run[i] = l_run[i] * alpha + rs;
            m_run[i] = mnew;
#pragma unroll
            for (int j = 0; j < 4; ++j) acc[j][i] *= alpha;
            pb[row * 32 + ln]      = f2bf(p0);
            pb[row * 32 + 16 + ln] = f2bf(p1);
        }
        // intra-wave LDS RAW: stores above feed the A-fragment load below
        asm volatile("s_wait_dscnt 0" ::: "memory");

        v16bf pf = load_frag_rowmajor(pb, 0, 32, lane);
#pragma unroll
        for (int j = 0; j < 4; ++j) {
            // V^T fragment: element(k = s-sb, n = hd) at (vh + sb)[n*SLEN + k]
            v16bf vf = load_frag_rowmajor(vh + sb, j * 16, SLEN, lane);
            acc[j] = __builtin_amdgcn_wmma_f32_16x16x32_bf16(false, pf, false, vf,
                                                             (short)0, acc[j], false, false);
        }
    }

    // cross-wave combine
    if (ln == 0) {
#pragma unroll
        for (int i = 0; i < 8; ++i) {
            int row = i + (g << 3);
            redM[w][row] = m_run[i];
            redL[w][row] = l_run[i];
        }
    }
#pragma unroll
    for (int j = 0; j < 4; ++j)
#pragma unroll
        for (int i = 0; i < 8; ++i) {
            int row = i + (g << 3);
            redO[w][row][j * 16 + ln] = acc[j][i];
        }
    __syncthreads();

    if (tid < 16) {
        int row = tid;
        float M = -3.0e38f;
        for (int ww = 0; ww < 8; ++ww) M = fmaxf(M, redM[ww][row]);
        float L = 0.f;
        for (int ww = 0; ww < 8; ++ww) {
            float sc = (redM[ww][row] > -1.0e37f) ? __expf(redM[ww][row] - M) : 0.f;
            scaleW[ww][row] = sc;
            L += sc * redL[ww][row];
        }
        Lrow[row] = L;
    }
    __syncthreads();

#pragma unroll
    for (int e = 0; e < 4; ++e) {
        int idx = tid * 4 + e;            // 1024 = 16 rows x 64 hd
        int row = idx >> 6, hd = idx & 63;
        float o = 0.f;
        for (int ww = 0; ww < 8; ++ww) o += scaleW[ww][row] * redO[ww][row][hd];
        o /= Lrow[row];
        int beam = row >> 2, t = row & 3;
        int b = cb * BEAMS + beam;
        ao[((long)(t * BATCH + b)) * EMBED + h * HDIM + hd] = f2bf(o);
    }
}

// ---------------------------------------------------------------------------
extern "C" void kernel_launch(void* const* d_in, const int* in_sizes, int n_in,
                              void* d_out, int out_size, void* d_ws, size_t ws_size,
                              hipStream_t stream) {
    const float* query = (const float*)d_in[0];
    const float* key   = (const float*)d_in[1];
    const unsigned char* kpm = (const unsigned char*)d_in[2];  // bool mask [B,S]
    const float* wq = (const float*)d_in[3];
    const float* bq = (const float*)d_in[4];
    const float* wk = (const float*)d_in[5];
    const float* bk = (const float*)d_in[6];
    const float* wv = (const float*)d_in[7];
    const float* bv = (const float*)d_in[8];
    const float* wo = (const float*)d_in[9];
    const float* bo = (const float*)d_in[10];

    char* ws = (char*)d_ws;
    const size_t MB = 1048576;
    unsigned short* keyc   = (unsigned short*)(ws);                 // 32 MiB [16384][1024]
    unsigned short* k_proj = (unsigned short*)(ws + 32  * MB);      // 32 MiB [cb][h][s][hd]
    unsigned short* vT     = (unsigned short*)(ws + 64  * MB);      // 32 MiB [cb][h][hd][s]
    unsigned short* wq_bf  = (unsigned short*)(ws + 96  * MB);      // 2 MiB each
    unsigned short* wk_bf  = (unsigned short*)(ws + 98  * MB);
    unsigned short* wv_bf  = (unsigned short*)(ws + 100 * MB);
    unsigned short* wo_bf  = (unsigned short*)(ws + 102 * MB);
    unsigned short* qraw   = (unsigned short*)(ws + 104 * MB);          // 256 KiB [128][1024]
    unsigned short* qtile  = (unsigned short*)(ws + 104 * MB + 262144); // 256 KiB [cb][h][16][hd]
    unsigned short* ao_bf  = (unsigned short*)(ws + 104 * MB + 524288); // 256 KiB [128][1024]

    dim3 blk(256);
    // one-time f32->bf16 conversions (bandwidth-bound)
    cvt_gather_key<<<8192, blk, 0, stream>>>(key, keyc);           // 16384*1024 / 2048
    cvt_plain<<<512, blk, 0, stream>>>(wk, wk_bf);                 // 1M / 2048
    cvt_plain<<<512, blk, 0, stream>>>(wv, wv_bf);
    cvt_plain<<<512, blk, 0, stream>>>(wq, wq_bf);
    cvt_plain<<<512, blk, 0, stream>>>(wo, wo_bf);
    cvt_plain<<<64,  blk, 0, stream>>>(query, qraw);               // 128K / 2048

    // K/V projections over dedup batch rows: M = 16384, N = 1024
    gemm_wmma<0><<<dim3(8, 128), blk, 0, stream>>>(keyc, wk_bf, bk, (void*)k_proj);
    gemm_wmma<3><<<dim3(8, 128), blk, 0, stream>>>(keyc, wv_bf, bv, (void*)vT);
    // Q projection: M = 128
    gemm_wmma<1><<<dim3(8, 1), blk, 0, stream>>>(qraw, wq_bf, bq, (void*)qtile);
    // Attention: one WG per (head, cache-batch)
    attn_wmma<<<dim3(HEADS, CBATCH), blk, 0, stream>>>(qtile, k_proj, vT, kpm, ao_bf);
    // Output projection -> f32 d_out
    gemm_wmma<2><<<dim3(8, 1), blk, 0, stream>>>(ao_bf, wo_bf, bo, d_out);
}
